// MultiHeadAttentionBlock_22333829939988
// MI455X (gfx1250) — compile-verified, matrix operand order checked by reading, not run on hardware
//
#include <hip/hip_runtime.h>
#include <cstdint>

typedef __attribute__((ext_vector_type(16))) _Float16 v16h;
typedef __attribute__((ext_vector_type(8)))  float    v8f;
typedef __attribute__((ext_vector_type(4)))  unsigned int v4u;
typedef __attribute__((ext_vector_type(8)))  int      v8i;
typedef __attribute__((ext_vector_type(4)))  int      v4i;

#define B_  2
#define S_  2048
#define D_  1024
#define H_  16
#define DK_ 64
#define KTILE 64
#define LDP 72   // padded LDS row stride in halves (144B: 16B-aligned, conflict-free)

union FragH { v16h v; _Float16 h[16]; uint4 q[2]; };
union FragF { v8f  v; float    f[8];  };
union H4   { uint2 u; _Float16 h[4]; };

__device__ inline v8f zero8() {
  FragF z;
#pragma unroll
  for (int i = 0; i < 8; ++i) z.f[i] = 0.0f;
  return z.v;
}

// 16x32 f16 fragment load (A: lane->row; B when source is "B^T row-major").
// ISA 7.12.2: lanes 0-15: row=lane, halves {K0..7,K16..23};
//             lanes 16-31: row=lane-16, halves {K8..15,K24..31}.
__device__ inline v16h frag_load(const _Float16* __restrict__ p, int ld) {
  const int lane = threadIdx.x & 31;
  const int r    = lane & 15;
  const int hs   = lane >> 4;
  const _Float16* row = p + r * ld + hs * 8;
  FragH f;
  f.q[0] = *reinterpret_cast<const uint4*>(row);
  f.q[1] = *reinterpret_cast<const uint4*>(row + 16);
  return f.v;
}

__device__ inline v8f wmma16(v16h a, v16h b, v8f c) {
  return __builtin_amdgcn_wmma_f32_16x16x32_f16(
      false, a, false, b, (short)0, c, false, false);
}

__device__ inline float red_max16(float x) {
  x = fmaxf(x, __shfl_xor(x, 1, 32));
  x = fmaxf(x, __shfl_xor(x, 2, 32));
  x = fmaxf(x, __shfl_xor(x, 4, 32));
  x = fmaxf(x, __shfl_xor(x, 8, 32));
  return x;
}
__device__ inline float red_sum16(float x) {
  x += __shfl_xor(x, 1, 32);
  x += __shfl_xor(x, 2, 32);
  x += __shfl_xor(x, 4, 32);
  x += __shfl_xor(x, 8, 32);
  return x;
}

// ---------------------------------------------------------------------------
// QKV projection GEMM: out[m, j] = sum_k A[m,k] * W[j,k] + bias[j], A is f32.
// MODE 0: f16 -> [B,H,S,DK]; MODE 1: f16 -> [B,H,DK,S] (V transposed).
// 64x64 tile, 4 waves, K-step 64, register-prefetch software pipeline.
// ---------------------------------------------------------------------------
template <int MODE>
__global__ __launch_bounds__(128) void proj_gemm_kernel(
    const float* __restrict__ Ap, const float* __restrict__ W,
    const float* __restrict__ bias, _Float16* __restrict__ outp) {
  __shared__ __align__(16) _Float16 At[64 * LDP];
  __shared__ __align__(16) _Float16 Bt[64 * LDP];

  const int t    = threadIdx.x;
  const int wave = t >> 5;
  const int lane = t & 31;
  const int m0   = blockIdx.x * 64;
  const int n0   = blockIdx.y * 64;
  const float* Ab = Ap + (size_t)m0 * D_;
  const float* Wb = W + (size_t)n0 * D_;

  v8f acc[4];
#pragma unroll
  for (int i = 0; i < 4; ++i) acc[i] = zero8();

  float4 ra[8], rb[8];
  auto fetch = [&](int k0) {
#pragma unroll
    for (int i = 0; i < 8; ++i) {
      const int idx = t + i * 128, row = idx >> 4, c4 = idx & 15;
      ra[i] = *reinterpret_cast<const float4*>(Ab + (size_t)row * D_ + k0 + c4 * 4);
      rb[i] = *reinterpret_cast<const float4*>(Wb + (size_t)row * D_ + k0 + c4 * 4);
    }
  };
  auto stage = [&]() {
#pragma unroll
    for (int i = 0; i < 8; ++i) {
      const int idx = t + i * 128, row = idx >> 4, c4 = idx & 15;
      H4 pa, pb;
      pa.h[0] = (_Float16)ra[i].x; pa.h[1] = (_Float16)ra[i].y;
      pa.h[2] = (_Float16)ra[i].z; pa.h[3] = (_Float16)ra[i].w;
      pb.h[0] = (_Float16)rb[i].x; pb.h[1] = (_Float16)rb[i].y;
      pb.h[2] = (_Float16)rb[i].z; pb.h[3] = (_Float16)rb[i].w;
      *reinterpret_cast<uint2*>(&At[row * LDP + c4 * 4]) = pa.u;
      *reinterpret_cast<uint2*>(&Bt[row * LDP + c4 * 4]) = pb.u;
    }
  };

  fetch(0);
  const int NK = D_ / KTILE;  // 16
  for (int kb = 0; kb < NK; ++kb) {
    stage();
    __syncthreads();
    if (kb + 1 < NK) fetch((kb + 1) * KTILE);  // overlap with ds_load+wmma below
    const v16h b0 = frag_load(Bt + wave * 16 * LDP,      LDP);
    const v16h b1 = frag_load(Bt + wave * 16 * LDP + 32, LDP);
#pragma unroll
    for (int ms = 0; ms < 4; ++ms) {
      acc[ms] = wmma16(frag_load(At + ms * 16 * LDP,      LDP), b0, acc[ms]);
      acc[ms] = wmma16(frag_load(At + ms * 16 * LDP + 32, LDP), b1, acc[ms]);
    }
    __syncthreads();
  }

  const int n  = lane & 15;
  const int mb = (lane >> 4) * 8;
  const int j  = n0 + wave * 16 + n;
  const float bj = bias[j];
#pragma unroll
  for (int ms = 0; ms < 4; ++ms) {
    FragF c; c.v = acc[ms];
#pragma unroll
    for (int r = 0; r < 8; ++r) {
      const int   m   = m0 + ms * 16 + mb + r;
      const float val = c.f[r] + bj;
      const int b = m >> 11, s = m & (S_ - 1);
      const int h = j >> 6,  dk = j & 63;
      if constexpr (MODE == 0)
        outp[(((size_t)(b * H_ + h)) * S_ + s) * DK_ + dk] = (_Float16)val;
      else
        outp[(((size_t)(b * H_ + h)) * DK_ + dk) * S_ + s] = (_Float16)val;
    }
  }
}

// ---------------------------------------------------------------------------
// Output projection: out[m,j] = sum_k Cx[m,k]*Wo[j,k] + bo[j], Cx is f16.
// A tile staged by the Tensor Data Mover (tensor_load_to_lds) with LDS
// padding reproducing the LDP=72 layout; double-buffered, TENSORcnt-paced.
// ---------------------------------------------------------------------------
__global__ __launch_bounds__(128) void out_proj_tdm_kernel(
    const _Float16* __restrict__ Cx, const float* __restrict__ W,
    const float* __restrict__ bias, float* __restrict__ outp) {
  __shared__ __align__(16) _Float16 At[2][64 * LDP];
  __shared__ __align__(16) _Float16 Bt[64 * LDP];

  const int t    = threadIdx.x;
  const int wave = t >> 5;
  const int lane = t & 31;
  const int m0   = blockIdx.x * 64;
  const int n0   = blockIdx.y * 64;
  const float* Wb = W + (size_t)n0 * D_;

  v8f acc[4];
#pragma unroll
  for (int i = 0; i < 4; ++i) acc[i] = zero8();

  // Issue one TDM load of a 64x64 f16 tile of Cx into At[buf], with 4 pad
  // DWORDs after every 32 data DWORDs (row of 64 halves) -> stride LDP=72.
  auto tdm_issue = [&](int buf, int k0) {
    const unsigned lds_off = (unsigned)(uintptr_t)(void*)&At[buf][0];
    const unsigned long long ga =
        (unsigned long long)(uintptr_t)(const void*)Cx +
        ((size_t)m0 * D_ + k0) * 2ull;
    v4u g0;
    g0.x = 1u;                                     // count=1 valid descriptor
    g0.y = lds_off;                                // D#.lds_addr (bytes)
    g0.z = (unsigned)(ga & 0xFFFFFFFFu);           // global_addr[31:0]
    g0.w = (unsigned)((ga >> 32) & 0x1FFFFFFu)     // global_addr[56:32]
           | (2u << 30);                           // type=2 ("image")
    v8i g1;
    g1[0] = (int)((1u << 16)        // data_size = 1 -> 2 bytes
                | (1u << 20)        // pad_enable
                | (4u << 22)        // pad_interval: 2^(4+1)=32 DWORDs
                | (3u << 25));      // pad_amount: 3+1 = 4 DWORDs
    g1[1] = (int)(((unsigned)D_ & 0xFFFFu) << 16);         // tensor_dim0 lo16
    g1[2] = (int)((((unsigned)(B_ * S_)) & 0xFFFFu) << 16); // dim0 hi=0 | tensor_dim1 lo16 (4096)
    g1[3] = (int)(64u << 16);       // tensor_dim1 hi=0 | tile_dim0 = 64
    g1[4] = (int)(64u);             // tile_dim1 = 64 | tile_dim2 = 0
    g1[5] = (int)((unsigned)D_);    // tensor_dim0_stride lo32 = 1024
    g1[6] = 0;                      // stride0 hi16 | stride1 lo16 (unused, 2D)
    g1[7] = 0;
    const v4i gz = {0, 0, 0, 0};
#if defined(__clang_major__) && (__clang_major__ >= 23)
    const v8i gz8 = {0, 0, 0, 0, 0, 0, 0, 0};
    __builtin_amdgcn_tensor_load_to_lds(g0, g1, gz, gz, gz8, 0);
#else
    __builtin_amdgcn_tensor_load_to_lds(g0, g1, gz, gz, 0);
#endif
  };

  float4 rb[8];
  auto fetchB = [&](int k0) {
#pragma unroll
    for (int i = 0; i < 8; ++i) {
      const int idx = t + i * 128, row = idx >> 4, c4 = idx & 15;
      rb[i] = *reinterpret_cast<const float4*>(Wb + (size_t)row * D_ + k0 + c4 * 4);
    }
  };
  auto stageB = [&]() {
#pragma unroll
    for (int i = 0; i < 8; ++i) {
      const int idx = t + i * 128, row = idx >> 4, c4 = idx & 15;
      H4 pk;
      pk.h[0] = (_Float16)rb[i].x; pk.h[1] = (_Float16)rb[i].y;
      pk.h[2] = (_Float16)rb[i].z; pk.h[3] = (_Float16)rb[i].w;
      *reinterpret_cast<uint2*>(&Bt[row * LDP + c4 * 4]) = pk.u;
    }
  };

  fetchB(0);
  if (wave == 0) tdm_issue(0, 0);
  const int NK = D_ / KTILE;  // 16
  for (int kb = 0; kb < NK; ++kb) {
    // Prefetch slab kb+1 into the other LDS buffer (WAR-safe: everyone passed
    // the previous iteration's tail barrier before we get here).
    if (kb + 1 < NK && wave == 0) tdm_issue((kb + 1) & 1, (kb + 1) * KTILE);
    stageB();
    if (wave == 0) {
      if (kb + 1 < NK) __builtin_amdgcn_s_wait_tensorcnt(1);  // slab kb landed
      else             __builtin_amdgcn_s_wait_tensorcnt(0);
    }
    __syncthreads();
    if (kb + 1 < NK) fetchB((kb + 1) * KTILE);  // overlap with compute
    const _Float16* Ac = &At[kb & 1][0];
    const v16h b0 = frag_load(Bt + wave * 16 * LDP,      LDP);
    const v16h b1 = frag_load(Bt + wave * 16 * LDP + 32, LDP);
#pragma unroll
    for (int ms = 0; ms < 4; ++ms) {
      acc[ms] = wmma16(frag_load(Ac + ms * 16 * LDP,      LDP), b0, acc[ms]);
      acc[ms] = wmma16(frag_load(Ac + ms * 16 * LDP + 32, LDP), b1, acc[ms]);
    }
    __syncthreads();
  }

  const int n  = lane & 15;
  const int mb = (lane >> 4) * 8;
  const int j  = n0 + wave * 16 + n;
  const float bj = bias[j];
#pragma unroll
  for (int ms = 0; ms < 4; ++ms) {
    FragF c; c.v = acc[ms];
#pragma unroll
    for (int r = 0; r < 8; ++r) {
      const int m = m0 + ms * 16 + mb + r;
      outp[(size_t)m * D_ + j] = c.f[r] + bj;
    }
  }
}

// ---------------------------------------------------------------------------
// Causal flash attention, 64-key blocks.  One 16-row query tile per wave.
// Q/K: [BH, S, 64] f16   Vt: [BH, 64, S] f16   Ctx: row-major [B*S, D] f16.
// ---------------------------------------------------------------------------
__global__ __launch_bounds__(128) void attn_kernel(
    const _Float16* __restrict__ Qh, const _Float16* __restrict__ Kh,
    const _Float16* __restrict__ Vt, _Float16* __restrict__ Ctx) {
  __shared__ __align__(16) _Float16 Plds[4][16 * LDP];

  const int t    = threadIdx.x;
  const int wave = t >> 5;
  const int lane = t & 31;
  const int bh   = blockIdx.y;
  const int q0   = blockIdx.x * 64 + wave * 16;  // wave-uniform
  const int n    = lane & 15;
  const int mb   = (lane >> 4) * 8;

  const _Float16* Qb = Qh + (size_t)bh * S_ * DK_;
  const _Float16* Kb = Kh + (size_t)bh * S_ * DK_;
  const _Float16* Vb = Vt + (size_t)bh * DK_ * S_;
  _Float16* Pw = &Plds[wave][0];

  // Q fragments, pre-scaled by 1/sqrt(DK)
  FragH qlo, qhi;
  qlo.v = frag_load(Qb + (size_t)q0 * DK_ + 0,  DK_);
  qhi.v = frag_load(Qb + (size_t)q0 * DK_ + 32, DK_);
#pragma unroll
  for (int i = 0; i < 16; ++i) {
    qlo.h[i] = qlo.h[i] * (_Float16)0.125f;
    qhi.h[i] = qhi.h[i] * (_Float16)0.125f;
  }

  FragF ca[4];
#pragma unroll
  for (int i = 0; i < 4; ++i) ca[i].v = zero8();
  float mrow[8], lrow[8];
#pragma unroll
  for (int r = 0; r < 8; ++r) { mrow[r] = -3.0e38f; lrow[r] = 0.0f; }

  const int kbmax = (q0 + 15) >> 6;  // wave-uniform trip count
  for (int kb = 0; kb <= kbmax; ++kb) {
    const int key0 = kb * 64;

    // scores: 16 queries x 64 keys -> 8 WMMAs (K-frags straight from global)
    FragF sf[4];
#pragma unroll
    for (int qn = 0; qn < 4; ++qn) {
      v8f s = zero8();
      s = wmma16(qlo.v, frag_load(Kb + (size_t)(key0 + qn * 16) * DK_ + 0,  DK_), s);
      s = wmma16(qhi.v, frag_load(Kb + (size_t)(key0 + qn * 16) * DK_ + 32, DK_), s);
      sf[qn].v = s;
    }

#pragma unroll
    for (int r = 0; r < 8; ++r) {
      const int qi = q0 + mb + r;
      const float a0 = (key0 + n      <= qi) ? sf[0].f[r] : -3.0e38f;
      const float a1 = (key0 + 16 + n <= qi) ? sf[1].f[r] : -3.0e38f;
      const float a2 = (key0 + 32 + n <= qi) ? sf[2].f[r] : -3.0e38f;
      const float a3 = (key0 + 48 + n <= qi) ? sf[3].f[r] : -3.0e38f;

      const float rm  = red_max16(fmaxf(fmaxf(a0, a1), fmaxf(a2, a3)));
      const float nm  = fmaxf(mrow[r], rm);
      const float scl = __expf(mrow[r] - nm);
      const float p0  = __expf(a0 - nm);
      const float p1  = __expf(a1 - nm);
      const float p2  = __expf(a2 - nm);
      const float p3  = __expf(a3 - nm);
      const float rs  = red_sum16((p0 + p1) + (p2 + p3));
      lrow[r] = lrow[r] * scl + rs;
      mrow[r] = nm;
#pragma unroll
      for (int d = 0; d < 4; ++d) ca[d].f[r] *= scl;
      Pw[(mb + r) * LDP + n]      = (_Float16)p0;
      Pw[(mb + r) * LDP + 16 + n] = (_Float16)p1;
      Pw[(mb + r) * LDP + 32 + n] = (_Float16)p2;
      Pw[(mb + r) * LDP + 48 + n] = (_Float16)p3;
    }

    // ctx += P(16x64) @ V(64x64) -> 8 WMMAs.  P store/load is same-wave LDS
    // (in-order), no barrier; per-wave trip counts may differ.
    const v16h pf0 = frag_load(Pw,      LDP);
    const v16h pf1 = frag_load(Pw + 32, LDP);
#pragma unroll
    for (int d = 0; d < 4; ++d) {
      ca[d].v = wmma16(pf0, frag_load(Vb + (size_t)(d * 16) * S_ + key0,      S_), ca[d].v);
      ca[d].v = wmma16(pf1, frag_load(Vb + (size_t)(d * 16) * S_ + key0 + 32, S_), ca[d].v);
    }
  }

  const int b = bh >> 4, h = bh & 15;
#pragma unroll
  for (int r = 0; r < 8; ++r) {
    const float inv = 1.0f / lrow[r];
    const int   qi  = q0 + mb + r;
    const size_t base = ((size_t)b * S_ + qi) * D_ + h * DK_;
#pragma unroll
    for (int d = 0; d < 4; ++d)
      Ctx[base + d * 16 + n] = (_Float16)(ca[d].f[r] * inv);
  }
}

extern "C" void kernel_launch(void* const* d_in, const int* in_sizes, int n_in,
                              void* d_out, int out_size, void* d_ws, size_t ws_size,
                              hipStream_t stream) {
  (void)in_sizes; (void)n_in; (void)out_size; (void)ws_size;
  const float* q  = (const float*)d_in[0];
  const float* k  = (const float*)d_in[1];
  const float* v  = (const float*)d_in[2];
  // d_in[3]: causal tril mask -- causality applied analytically.
  const float* Wq = (const float*)d_in[4];
  const float* bq = (const float*)d_in[5];
  const float* Wk = (const float*)d_in[6];
  const float* bk = (const float*)d_in[7];
  const float* Wv = (const float*)d_in[8];
  const float* bv = (const float*)d_in[9];
  const float* Wo = (const float*)d_in[10];
  const float* bo = (const float*)d_in[11];
  float* out = (float*)d_out;

  _Float16* ws = (_Float16*)d_ws;
  const size_t SEG = (size_t)B_ * H_ * S_ * DK_;  // 4,194,304 halves = 8 MB
  _Float16* Qh = ws;
  _Float16* Kh = ws + SEG;
  _Float16* Vt = ws + 2 * SEG;
  _Float16* Cx = ws + 3 * SEG;

  const dim3 blk(128, 1, 1);
  const dim3 gproj(64, 16, 1);  // M/64 x N/64
  const dim3 gattn(32, 32, 1);  // S/64 x B*H

  proj_gemm_kernel<0><<<gproj, blk, 0, stream>>>(q, Wq, bq, Qh);
  proj_gemm_kernel<0><<<gproj, blk, 0, stream>>>(k, Wk, bk, Kh);
  proj_gemm_kernel<1><<<gproj, blk, 0, stream>>>(v, Wv, bv, Vt);
  attn_kernel<<<gattn, blk, 0, stream>>>(Qh, Kh, Vt, Cx);
  out_proj_tdm_kernel<<<gproj, blk, 0, stream>>>(Cx, Wo, bo, out);
}
